// QuasarMoE_80771154968637
// MI455X (gfx1250) — compile-verified, matrix operand order checked by reading, not run on hardware
//
#include <hip/hip_runtime.h>
#include <hip/hip_bf16.h>
#include <math.h>

// ---------------------------------------------------------------------------
// MoE (sigmoid top-2 routing + shared expert, SwiGLU FFNs) for MI455X/gfx1250.
// bf16 WMMA grouped GEMM; double-buffered LDS; TDM async loads for bf16
// activation tiles overlapped with WMMA compute.
// ---------------------------------------------------------------------------

typedef __attribute__((ext_vector_type(16))) __bf16 v16bf;
typedef __attribute__((ext_vector_type(8)))  __bf16 v8bf;
typedef __attribute__((ext_vector_type(8)))  float  v8f;
typedef __attribute__((ext_vector_type(4)))  unsigned int u32x4;
typedef __attribute__((ext_vector_type(4)))  int i32x4;
typedef __attribute__((ext_vector_type(8)))  int i32x8;

#define BM   64            // rows (tokens) per block tile
#define BN   128           // cols per block tile
#define BK   32            // K-depth per WMMA step (bf16 16x16x32)
#define PAD  8
#define LSTR (BK + PAD)    // LDS row stride in elements (40 -> 80B, 16B aligned)
#define ASZ  (BM * LSTR)   // one A buffer, elements
#define BSZ  (BN * LSTR)   // one B buffer, elements

// Load one 16x32 bf16 fragment (A or N-major B) from LDS per the CDNA5 ISA
// layout: lanes 0-15 hold K={0..7,16..23}, lanes 16-31 hold K={8..15,24..31}.
__device__ __forceinline__ v16bf ldfrag(const __bf16* base, int lane) {
  const int half = lane >> 4;
  const int r    = lane & 15;
  const __bf16* p = base + r * LSTR + half * 8;
  v8bf lo = *(const v8bf*)(p);
  v8bf hi = *(const v8bf*)(p + 16);
  return __builtin_shufflevector(lo, hi,
      0,1,2,3,4,5,6,7,8,9,10,11,12,13,14,15);
}

// Fast sigmoid: v_exp_f32 + raw v_rcp_f32 (no IEEE div fixup sequence).
__device__ __forceinline__ float sigmoid_f(float v) {
  return __builtin_amdgcn_rcpf(1.0f + __expf(-v));
}

// Cooperative load of a BKxBN f32 weight tile into LDS as N-major bf16.
// Thread t owns an 8(K) x 2(N) sub-column: 8 coalesced float2 loads, then two
// packed v8bf (ds_store_b128) stores -- no scalar LDS traffic.
__device__ __forceinline__ void load_wtile(const float* __restrict__ w,
                                           __bf16* __restrict__ Bs,
                                           int k0, int n0, int N, int tid) {
  const int kq = (tid & 3) * 8;        // K sub-block base: 0,8,16,24
  const int nc = (tid >> 2) * 2;       // N pair base: 0..126
  float2 v[8];
#pragma unroll
  for (int i = 0; i < 8; ++i)
    v[i] = *(const float2*)(w + (size_t)(k0 + kq + i) * N + n0 + nc);
  v8bf c0, c1;
#pragma unroll
  for (int i = 0; i < 8; ++i) { c0[i] = (__bf16)v[i].x; c1[i] = (__bf16)v[i].y; }
  *(v8bf*)(Bs + (nc + 0) * LSTR + kq) = c0;
  *(v8bf*)(Bs + (nc + 1) * LSTR + kq) = c1;
}

// ---------------------------------------------------------------------------
// Tensor Data Mover: 2D bf16 tile (tile_rows x tile_cols) global -> LDS with
// pad_interval=16 DWORDs / pad_amount=4 DWORDs to produce the LSTR=40 element
// padded row stride. OOB rows (>= tensor_rows) are zero-filled by hardware.
// ---------------------------------------------------------------------------
__device__ __forceinline__ void tdm_load_2d_bf16(
    unsigned lds_addr, const void* gptr, unsigned tensor_rows,
    unsigned tile_cols, unsigned tile_rows, unsigned row_stride_elems)
{
  const unsigned long long ga = (unsigned long long)(uintptr_t)gptr;
  u32x4 g0;
  g0[0] = 1u;                                          // count=1, user mode
  g0[1] = lds_addr;                                    // LDS byte address
  g0[2] = (unsigned)(ga & 0xFFFFFFFFu);                // global addr [31:0]
  g0[3] = (unsigned)((ga >> 32) & 0x1FFFFFFu)          // global addr [56:32]
        | (2u << 30);                                  // type = image

  i32x8 g1;
  const unsigned d0 = (1u << 16)                       // data_size = 2 bytes
                    | (1u << 20)                       // pad_enable
                    | (3u << 22)                       // pad_interval: 16 DW
                    | (3u << 25);                      // pad_amount: 4 DW
  g1[0] = (int)d0;
  g1[1] = (int)((tile_cols & 0xFFFFu) << 16);          // tensor_dim0 lo16
  g1[2] = (int)((tile_cols >> 16) |
                ((tensor_rows & 0xFFFFu) << 16));      // dim0 hi | dim1 lo
  g1[3] = (int)((tensor_rows >> 16) |
                ((tile_cols & 0xFFFFu) << 16));        // dim1 hi | tile_dim0
  g1[4] = (int)(tile_rows & 0xFFFFu);                  // tile_dim1 (dim2=0)
  g1[5] = (int)row_stride_elems;                       // dim0_stride lo32
  g1[6] = 0;
  g1[7] = 0;

  const i32x4 z4 = {0, 0, 0, 0};
#if __has_include(<hip/amd_detail/amd_gfx1250_TDM.h>)
  const i32x8 z8 = {0, 0, 0, 0, 0, 0, 0, 0};
  __builtin_amdgcn_tensor_load_to_lds(g0, g1, z4, z4, z8, 0);
#else
  __builtin_amdgcn_tensor_load_to_lds(g0, g1, z4, z4, 0);
#endif
}

// ---------------------------------------------------------------------------
// f32 -> bf16 bulk convert (for the activation matrix x)
// ---------------------------------------------------------------------------
__global__ __launch_bounds__(256) void cvt_bf16_kernel(
    const float* __restrict__ in, __bf16* __restrict__ out, int n8)
{
  const int i = blockIdx.x * 256 + threadIdx.x;
  if (i >= n8) return;
  const float4* s = (const float4*)in + (size_t)i * 2;
  const float4 a = s[0], b = s[1];
  v8bf v;
  v[0] = (__bf16)a.x; v[1] = (__bf16)a.y; v[2] = (__bf16)a.z; v[3] = (__bf16)a.w;
  v[4] = (__bf16)b.x; v[5] = (__bf16)b.y; v[6] = (__bf16)b.z; v[7] = (__bf16)b.w;
  ((v8bf*)out)[i] = v;
}

// ---------------------------------------------------------------------------
// Router: one wave per token. logits = x@Wr + br ; select top2 on logits+eb;
// gates = sigmoid(logits[idx]) / sum. Atomic-append (token, gate) per expert.
// ---------------------------------------------------------------------------
__global__ __launch_bounds__(256) void moe_router_kernel(
    const float* __restrict__ x, const float* __restrict__ Wr,
    const float* __restrict__ br, const float* __restrict__ eb,
    int* __restrict__ cnt, int* __restrict__ tok, float* __restrict__ gates,
    int H, int cap, int ntok)
{
  const int t    = (blockIdx.x * blockDim.x + threadIdx.x) >> 5;
  const int lane = threadIdx.x & 31;
  if (t >= ntok) return;

  float acc[8];
#pragma unroll
  for (int e = 0; e < 8; ++e) acc[e] = 0.0f;

  const float* xr = x + (size_t)t * H;
  for (int k = lane; k < H; k += 32) {
    const float xv = xr[k];
    const float* w = Wr + k * 8;
#pragma unroll
    for (int e = 0; e < 8; ++e) acc[e] += xv * w[e];
  }
#pragma unroll
  for (int e = 0; e < 8; ++e) {
#pragma unroll
    for (int m = 16; m >= 1; m >>= 1) acc[e] += __shfl_xor(acc[e], m, 32);
  }

  if (lane == 0) {
    float logit[8], sel[8];
#pragma unroll
    for (int e = 0; e < 8; ++e) { logit[e] = acc[e] + br[e]; sel[e] = logit[e] + eb[e]; }
    int i0 = 0;
#pragma unroll
    for (int e = 1; e < 8; ++e) if (sel[e] > sel[i0]) i0 = e;
    int i1 = (i0 == 0) ? 1 : 0;
#pragma unroll
    for (int e = 0; e < 8; ++e) if (e != i0 && sel[e] > sel[i1]) i1 = e;

    const float s0 = sigmoid_f(logit[i0]);
    const float s1 = sigmoid_f(logit[i1]);
    const float rdn = __builtin_amdgcn_rcpf(fmaxf(s0 + s1, 1e-12f));

    int p0 = atomicAdd(&cnt[i0], 1);
    tok[i0 * cap + p0]   = t;
    gates[i0 * cap + p0] = s0 * rdn;
    int p1 = atomicAdd(&cnt[i1], 1);
    tok[i1 * cap + p1]   = t;
    gates[i1 * cap + p1] = s1 * rdn;
  }
}

// ---------------------------------------------------------------------------
// Up-projection (grouped): h[i,:] = silu(xb[tok[i]]@w1 + b1) * (xb[tok[i]]@w3 + b3)
// Double-buffered LDS: tile k+1 global loads are issued before tile k compute.
// 8 waves: 2 (wave_m) x 4 (wave_n); each wave = 2x2 WMMA tiles of 16x16.
// ---------------------------------------------------------------------------
__global__ __launch_bounds__(256) void moe_up_kernel(
    const __bf16* __restrict__ xb, const int* __restrict__ tok,
    const int* __restrict__ cntp, int nrows_fixed,
    const float* __restrict__ w1, const float* __restrict__ b1,
    const float* __restrict__ w3, const float* __restrict__ b3,
    __bf16* __restrict__ hbuf, int H, int I)
{
  const int nrows = cntp ? cntp[0] : nrows_fixed;
  const int row0  = blockIdx.y * BM;
  if (row0 >= nrows) return;
  const int n0 = blockIdx.x * BN;

  const int tid  = threadIdx.x;
  const int lane = tid & 31;
  const int wv   = tid >> 5;
  const int wm   = wv & 1;
  const int wn   = wv >> 1;

  __shared__ __align__(16) __bf16 As [2 * ASZ];
  __shared__ __align__(16) __bf16 Bs1[2 * BSZ];
  __shared__ __align__(16) __bf16 Bs3[2 * BSZ];

  // A-tile mapping: 4 threads per row, 8 bf16 (16B) each
  const int arow  = tid >> 2;
  const int acol  = (tid & 3) * 8;
  const int grow  = row0 + arow;
  const bool rv   = grow < nrows;
  const int token = rv ? (tok ? tok[grow] : grow) : 0;
  const __bf16* asrc = xb + (size_t)token * H;

  const v8f vz = {};
  v8f acc1[2][2], acc3[2][2];
#pragma unroll
  for (int i = 0; i < 2; ++i)
#pragma unroll
    for (int j = 0; j < 2; ++j) { acc1[i][j] = vz; acc3[i][j] = vz; }

  // Prologue: tile 0 into buffer 0
  {
    v8bf u = {};
    if (rv) u = *(const v8bf*)(asrc + acol);
    *(v8bf*)(As + arow * LSTR + acol) = u;
    load_wtile(w1, Bs1, 0, n0, I, tid);
    load_wtile(w3, Bs3, 0, n0, I, tid);
  }
  __syncthreads();

  const int KT = H / BK;
  for (int kt = 0; kt < KT; ++kt) {
    const int cur = kt & 1;
    const int nxt = cur ^ 1;
    if (kt + 1 < KT) {                  // prefetch tile k+1 into other buffer
      const int kn = (kt + 1) * BK;
      v8bf u = {};
      if (rv) u = *(const v8bf*)(asrc + kn + acol);
      *(v8bf*)(As + nxt * ASZ + arow * LSTR + acol) = u;
      load_wtile(w1, Bs1 + nxt * BSZ, kn, n0, I, tid);
      load_wtile(w3, Bs3 + nxt * BSZ, kn, n0, I, tid);
    }

    v16bf af[2], f1[2], f3[2];
#pragma unroll
    for (int tm = 0; tm < 2; ++tm)
      af[tm] = ldfrag(As + cur * ASZ + (wm * 32 + tm * 16) * LSTR, lane);
#pragma unroll
    for (int tn = 0; tn < 2; ++tn) {
      f1[tn] = ldfrag(Bs1 + cur * BSZ + (wn * 32 + tn * 16) * LSTR, lane);
      f3[tn] = ldfrag(Bs3 + cur * BSZ + (wn * 32 + tn * 16) * LSTR, lane);
    }
#pragma unroll
    for (int tm = 0; tm < 2; ++tm)
#pragma unroll
      for (int tn = 0; tn < 2; ++tn) {
        acc1[tm][tn] = __builtin_amdgcn_wmma_f32_16x16x32_bf16(
            false, af[tm], false, f1[tn], (short)0, acc1[tm][tn], false, false);
        acc3[tm][tn] = __builtin_amdgcn_wmma_f32_16x16x32_bf16(
            false, af[tm], false, f3[tn], (short)0, acc3[tm][tn], false, false);
      }
    __syncthreads();   // next-buffer stores visible; cur buffer free to overwrite
  }

  // Epilogue: fused SwiGLU, bf16 store of intermediate
  const int cn   = lane & 15;
  const int moff = (lane >> 4) * 8;
#pragma unroll
  for (int tm = 0; tm < 2; ++tm) {
#pragma unroll
    for (int tn = 0; tn < 2; ++tn) {
      const int col = n0 + wn * 32 + tn * 16 + cn;
      const float bb1 = b1[col], bb3 = b3[col];
#pragma unroll
      for (int r = 0; r < 8; ++r) {
        const int row = row0 + wm * 32 + tm * 16 + moff + r;
        if (row < nrows) {
          const float v1 = acc1[tm][tn][r] + bb1;
          const float v3 = acc3[tm][tn][r] + bb3;
          const float hv = (v1 * sigmoid_f(v1)) * v3;
          hbuf[(size_t)row * I + col] = (__bf16)hv;
        }
      }
    }
  }
}

// ---------------------------------------------------------------------------
// Down-projection (grouped): out[tok[i]] (+)= gate[i] * (h[i]@w2 + b2)
// A-tile via async TDM (tensor_load_to_lds) double-buffered: the DMA for tile
// k+1 runs while tile k's WMMAs execute; s_wait_tensorcnt after the compute.
// ---------------------------------------------------------------------------
__global__ __launch_bounds__(256) void moe_down_kernel(
    const __bf16* __restrict__ hbuf, const int* __restrict__ tok,
    const float* __restrict__ gatep, const int* __restrict__ cntp, int nrows_fixed,
    const float* __restrict__ w2, const float* __restrict__ b2,
    const float* __restrict__ xbase, float* __restrict__ out,
    int accumulate, int I, int H)
{
  const int nrows = cntp ? cntp[0] : nrows_fixed;
  const int row0  = blockIdx.y * BM;
  if (row0 >= nrows) return;
  const int n0 = blockIdx.x * BN;

  const int tid  = threadIdx.x;
  const int lane = tid & 31;
  const int wv   = tid >> 5;
  const int wm   = wv & 1;
  const int wn   = wv >> 1;

  __shared__ __align__(16) __bf16 As[2 * ASZ];
  __shared__ __align__(16) __bf16 Bs[2 * BSZ];

  const unsigned lds_as    = (unsigned)(uintptr_t)(&As[0]);
  const unsigned trow_left = (unsigned)(nrows - row0);
  const __bf16*  hrow      = hbuf + (size_t)row0 * I;

  const v8f vz = {};
  v8f acc[2][2];
#pragma unroll
  for (int i = 0; i < 2; ++i)
#pragma unroll
    for (int j = 0; j < 2; ++j) acc[i][j] = vz;

  // Prologue: tile 0
  if (wv == 0)
    tdm_load_2d_bf16(lds_as, hrow, trow_left, (unsigned)BK, (unsigned)BM,
                     (unsigned)I);
  load_wtile(w2, Bs, 0, n0, H, tid);
  if (wv == 0) __builtin_amdgcn_s_wait_tensorcnt(0);
  __syncthreads();

  const int KT = I / BK;
  for (int kt = 0; kt < KT; ++kt) {
    const int cur = kt & 1;
    const int nxt = cur ^ 1;
    const bool pf = (kt + 1 < KT);
    if (pf) {                            // async prefetch of tile k+1
      if (wv == 0)
        tdm_load_2d_bf16(lds_as + (unsigned)(nxt * ASZ * 2),
                         hrow + (kt + 1) * BK, trow_left,
                         (unsigned)BK, (unsigned)BM, (unsigned)I);
      load_wtile(w2, Bs + nxt * BSZ, (kt + 1) * BK, n0, H, tid);
    }

    v16bf af[2], bf2[2];
#pragma unroll
    for (int tm = 0; tm < 2; ++tm)
      af[tm] = ldfrag(As + cur * ASZ + (wm * 32 + tm * 16) * LSTR, lane);
#pragma unroll
    for (int tn = 0; tn < 2; ++tn)
      bf2[tn] = ldfrag(Bs + cur * BSZ + (wn * 32 + tn * 16) * LSTR, lane);
#pragma unroll
    for (int tm = 0; tm < 2; ++tm)
#pragma unroll
      for (int tn = 0; tn < 2; ++tn)
        acc[tm][tn] = __builtin_amdgcn_wmma_f32_16x16x32_bf16(
            false, af[tm], false, bf2[tn], (short)0, acc[tm][tn], false, false);

    if (pf && wv == 0) __builtin_amdgcn_s_wait_tensorcnt(0);
    __syncthreads();
  }

  const int cn   = lane & 15;
  const int moff = (lane >> 4) * 8;
  if (accumulate) {
#pragma unroll
    for (int tm = 0; tm < 2; ++tm)
#pragma unroll
      for (int tn = 0; tn < 2; ++tn) {
        const int col = n0 + wn * 32 + tn * 16 + cn;
        const float bb2 = b2[col];
#pragma unroll
        for (int r = 0; r < 8; ++r) {
          const int lrow = row0 + wm * 32 + tm * 16 + moff + r;
          if (lrow < nrows) {
            const int   orow = tok[lrow];
            const float g    = gatep[lrow];
            float* dst = out + (size_t)orow * H + col;
            *dst += g * (acc[tm][tn][r] + bb2);
          }
        }
      }
  } else {
#pragma unroll
    for (int tm = 0; tm < 2; ++tm)
#pragma unroll
      for (int tn = 0; tn < 2; ++tn) {
        const int col = n0 + wn * 32 + tn * 16 + cn;
        const float bb2 = b2[col];
#pragma unroll
        for (int r = 0; r < 8; ++r) {
          const int lrow = row0 + wm * 32 + tm * 16 + moff + r;
          if (lrow < nrows) {
            out[(size_t)lrow * H + col] =
                xbase[(size_t)lrow * H + col] + acc[tm][tn][r] + bb2;
          }
        }
      }
  }
}

// ---------------------------------------------------------------------------
// Host-side launch
// ---------------------------------------------------------------------------
extern "C" void kernel_launch(void* const* d_in, const int* in_sizes, int n_in,
                              void* d_out, int out_size, void* d_ws, size_t ws_size,
                              hipStream_t stream) {
  (void)in_sizes; (void)n_in; (void)out_size; (void)ws_size;
  const int Bb = 4, S = 2048, H = 1024, I = 2048, E = 8;
  const int BS = Bb * S;         // 8192 tokens
  const int cap = BS;            // per-expert list capacity

  const float* x   = (const float*)d_in[0];
  const float* Wr  = (const float*)d_in[1];
  const float* br  = (const float*)d_in[2];
  const float* eb  = (const float*)d_in[3];
  const float* sw1 = (const float*)d_in[4];
  const float* sb1 = (const float*)d_in[5];
  const float* sw2 = (const float*)d_in[6];
  const float* sb2 = (const float*)d_in[7];
  const float* sw3 = (const float*)d_in[8];
  const float* sb3 = (const float*)d_in[9];
  const float* rw1 = (const float*)d_in[10];
  const float* rb1 = (const float*)d_in[11];
  const float* rw2 = (const float*)d_in[12];
  const float* rb2 = (const float*)d_in[13];
  const float* rw3 = (const float*)d_in[14];
  const float* rb3 = (const float*)d_in[15];
  float* out = (float*)d_out;

  char* ws = (char*)d_ws;
  size_t off = 0;
  int*    cnt   = (int*)(ws + off);  off += 256;                       // E ints
  int*    tok   = (int*)(ws + off);  off += (size_t)E * cap * 4;       // lists
  float*  gates = (float*)(ws + off); off += (size_t)E * cap * 4;
  __bf16* hbuf  = (__bf16*)(ws + off); off += (size_t)BS * I * 2;      // 32MB
  __bf16* xbf   = (__bf16*)(ws + off); off += (size_t)BS * H * 2;      // 16MB

  hipMemsetAsync(cnt, 0, E * sizeof(int), stream);
  cvt_bf16_kernel<<<(BS * H / 8 + 255) / 256, 256, 0, stream>>>(x, xbf, BS * H / 8);
  moe_router_kernel<<<BS / 8, 256, 0, stream>>>(x, Wr, br, eb, cnt, tok, gates,
                                                H, cap, BS);

  const dim3 gup(I / BN, BS / BM);   // (16, 128)
  const dim3 gdn(H / BN, BS / BM);   // (8, 128)

  // Shared expert: fresh-writes out = x + shared_ffn(x)
  moe_up_kernel<<<gup, 256, 0, stream>>>(xbf, nullptr, nullptr, BS,
                                         sw1, sb1, sw3, sb3, hbuf, H, I);
  moe_down_kernel<<<gdn, 256, 0, stream>>>(hbuf, nullptr, nullptr, nullptr, BS,
                                           sw2, sb2, x, out, 0, I, H);

  // Routed experts: grouped GEMM over gathered token lists, out += gate * ffn
  for (int e = 0; e < E; ++e) {
    moe_up_kernel<<<gup, 256, 0, stream>>>(
        xbf, tok + e * cap, cnt + e, 0,
        rw1 + (size_t)e * H * I, rb1 + (size_t)e * I,
        rw3 + (size_t)e * H * I, rb3 + (size_t)e * I, hbuf, H, I);
    moe_down_kernel<<<gdn, 256, 0, stream>>>(
        hbuf, tok + e * cap, gates + e * cap, cnt + e, 0,
        rw2 + (size_t)e * I * H, rb2 + (size_t)e * H, nullptr, out, 1, I, H);
  }
}
